// Recognition_RNN_36799279792190
// MI455X (gfx1250) — matrix operand order repaired and checked by reading, not compiled, not run on hardware
//
#include <hip/hip_runtime.h>

// RNN cell on gfx1250: fp32 GEMMs executed as 3-term bf16-split WMMA
// (hi*hi + hi*lo + lo*hi) with f32 accumulators.
// - pre-pass kernels split fp32 -> bf16 hi/lo planes (once per element)
// - GEMM stages tiles LDS-side with global_load_async_to_lds_b128 (ASYNCcnt),
//   double-buffered; inner loop is ds_load_b128 + v_wmma only.

typedef __attribute__((ext_vector_type(16))) __bf16 v16b;
typedef __attribute__((ext_vector_type(8)))  float  v8f;

#define BM 128
#define BN 128
#define BK 32
#define LSTR 40   // bf16 per LDS row: 32 data + 8 pad -> 80B rows, 16B-aligned, conflict-free b128

union Frag { int4 q[2]; v16b v; };

__device__ __forceinline__ unsigned bf16_hi_bits(float x) {
    unsigned u = __float_as_uint(x);
    return (u + 0x8000u) & 0xFFFF0000u;   // round-to-nearest bf16 kept in fp32 bit position
}

__device__ __forceinline__ void split_pair(float a, float b, unsigned& hi, unsigned& lo) {
    unsigned ha = bf16_hi_bits(a);
    unsigned hb = bf16_hi_bits(b);
    float ra = a - __uint_as_float(ha);
    float rb = b - __uint_as_float(hb);
    hi = (ha >> 16) | hb;                                   // a in [15:0], b in [31:16]
    lo = (bf16_hi_bits(ra) >> 16) | bf16_hi_bits(rb);
}

__device__ __forceinline__ unsigned lds_addr(const void* p) {
    // LDS aperture: addr[63:32] = shared_base, addr[31:0] = wave-relative LDS offset
    return (unsigned)(uintptr_t)p;
}

// Copy 32 contiguous bytes global -> LDS via the async engine (2x b128).
// GVS addressing: mem = SADDR + VADDR(u32) + INST_OFFSET; LDS = VDST + INST_OFFSET.
__device__ __forceinline__ void async_copy32B(unsigned lds, const void* base, unsigned goff) {
    asm volatile("global_load_async_to_lds_b128 %0, %1, %2 offset:0\n\t"
                 "global_load_async_to_lds_b128 %0, %1, %2 offset:16"
                 :: "v"(lds), "v"(goff), "s"(base) : "memory");
}

// ---------------- fp32 -> bf16 hi/lo split (elementwise pre-pass) ----------------
__global__ __launch_bounds__(256) void split_convert(
    const float* __restrict__ src, int src_ld,
    unsigned short* __restrict__ dhi, unsigned short* __restrict__ dlo,
    int dst_ld, int col_off, int rows, int cols)
{
    size_t nchunks = (size_t)rows * cols / 8;
    for (size_t i = (size_t)blockIdx.x * blockDim.x + threadIdx.x; i < nchunks;
         i += (size_t)gridDim.x * blockDim.x) {
        size_t e = i * 8;
        int r = (int)(e / cols);
        int c = (int)(e % cols);
        const float4* s4 = (const float4*)(src + (size_t)r * src_ld + c);
        float4 t0 = s4[0], t1 = s4[1];
        float f[8] = {t0.x, t0.y, t0.z, t0.w, t1.x, t1.y, t1.z, t1.w};
        unsigned hi[4], lo[4];
        #pragma unroll
        for (int q = 0; q < 4; ++q) split_pair(f[2*q], f[2*q+1], hi[q], lo[q]);
        size_t d = (size_t)r * dst_ld + col_off + c;
        *(int4*)(dhi + d) = make_int4((int)hi[0], (int)hi[1], (int)hi[2], (int)hi[3]);
        *(int4*)(dlo + d) = make_int4((int)lo[0], (int)lo[1], (int)lo[2], (int)lo[3]);
    }
}

// ---------------- WMMA GEMM: out = tanh?( A @ W^T + bias ) ----------------
// A planes: [Mtot, lda] bf16 hi/lo.  W planes: [Ntot, K] bf16 hi/lo (B[k][n] = W[n][k]).
template<bool DO_TANH>
__global__ __launch_bounds__(256) void wmma_gemm(
    const unsigned short* __restrict__ Ahi, const unsigned short* __restrict__ Alo, int lda,
    const unsigned short* __restrict__ Whi, const unsigned short* __restrict__ Wlo, int K,
    const float* __restrict__ bias, float* __restrict__ out, int ldout)
{
    __shared__ __align__(16) unsigned short sA[2][2][BM * LSTR];  // [buf][hi/lo]
    __shared__ __align__(16) unsigned short sB[2][2][BN * LSTR];

    const int tid  = threadIdx.x;
    const int lane = tid & 31;
    const int wid  = tid >> 5;   // 8 waves
    const int wm   = wid >> 2;   // 0..1 -> 64-row strip
    const int wn   = wid & 3;    // 0..3 -> 32-col strip

    const int m0 = blockIdx.x * BM;
    const int n0 = blockIdx.y * BN;

    // staging: 2 threads per row, 16 bf16 (32B) per plane each
    const int srow = tid >> 1;
    const int scol = (tid & 1) * 16;
    const unsigned ldsrow = (unsigned)(srow * LSTR + scol) * 2;  // byte offset within a plane

    v8f acc[4][2];
    #pragma unroll
    for (int i = 0; i < 4; ++i)
        #pragma unroll
        for (int j = 0; j < 2; ++j)
            acc[i][j] = {};

    const int hbsel     = (lane >= 16);
    const int arow_base = wm * 64 + (lane & 15);
    const int brow_base = wn * 32 + (lane & 15);

    const int T = K / BK;

    // async-stage tile t into buffer b (8 async instructions per thread)
    auto stage = [&](int t, int b) {
        unsigned k0 = (unsigned)(t * BK);
        unsigned aoff = (unsigned)(((unsigned)(m0 + srow) * (unsigned)lda + k0 + scol) * 2u);
        unsigned boff = (unsigned)(((unsigned)(n0 + srow) * (unsigned)K   + k0 + scol) * 2u);
        async_copy32B(lds_addr(&sA[b][0][0]) + ldsrow, Ahi, aoff);
        async_copy32B(lds_addr(&sA[b][1][0]) + ldsrow, Alo, aoff);
        async_copy32B(lds_addr(&sB[b][0][0]) + ldsrow, Whi, boff);
        async_copy32B(lds_addr(&sB[b][1][0]) + ldsrow, Wlo, boff);
    };

    stage(0, 0);

    for (int t = 0; t < T; ++t) {
        const int cur = t & 1;
        if (t + 1 < T) {
            stage(t + 1, cur ^ 1);
            asm volatile("s_wait_asynccnt 0x8" ::: "memory");  // tile t done; t+1 in flight
        } else {
            asm volatile("s_wait_asynccnt 0x0" ::: "memory");
        }
        __syncthreads();   // tile t visible to all waves

        // per-wave operand fragments (ISA VGPR layouts)
        Frag ah[4], al[4], bh[2], bl[2];
        #pragma unroll
        for (int i = 0; i < 4; ++i) {
            // A 16x32 bf16: lanes 0-15 hold K{0..7,16..23}, lanes 16-31 K{8..15,24..31}
            int base = (arow_base + i * 16) * LSTR + (hbsel ? 8 : 0);
            ah[i].q[0] = *(const int4*)&sA[cur][0][base];
            ah[i].q[1] = *(const int4*)&sA[cur][0][base + 16];
            al[i].q[0] = *(const int4*)&sA[cur][1][base];
            al[i].q[1] = *(const int4*)&sA[cur][1][base + 16];
        }
        #pragma unroll
        for (int j = 0; j < 2; ++j) {
            // B 32x16 bf16: column n = W row n; lanes 0-15 K{0..7,16..23}, lanes 16-31 K{8..15,24..31}
            int base = (brow_base + j * 16) * LSTR + (hbsel ? 8 : 0);
            bh[j].q[0] = *(const int4*)&sB[cur][0][base];
            bh[j].q[1] = *(const int4*)&sB[cur][0][base + 16];
            bl[j].q[0] = *(const int4*)&sB[cur][1][base];
            bl[j].q[1] = *(const int4*)&sB[cur][1][base + 16];
        }

        #pragma unroll
        for (int i = 0; i < 4; ++i)
            #pragma unroll
            for (int j = 0; j < 2; ++j) {
                acc[i][j] = __builtin_amdgcn_wmma_f32_16x16x32_bf16(
                    false, ah[i].v, false, bh[j].v, (short)0, acc[i][j], false, false);
                acc[i][j] = __builtin_amdgcn_wmma_f32_16x16x32_bf16(
                    false, ah[i].v, false, bl[j].v, (short)0, acc[i][j], false, false);
                acc[i][j] = __builtin_amdgcn_wmma_f32_16x16x32_bf16(
                    false, al[i].v, false, bh[j].v, (short)0, acc[i][j], false, false);
            }
        __syncthreads();   // all waves done reading buf[cur] before it is restaged
    }

    // epilogue: bias (+tanh); C/D layout: VGPR r -> M=r (+8 for lanes>=16), N=lane&15
    const int mrow0 = m0 + wm * 64 + (hbsel ? 8 : 0);
    const int col0  = n0 + wn * 32 + (lane & 15);
    #pragma unroll
    for (int i = 0; i < 4; ++i)
        #pragma unroll
        for (int j = 0; j < 2; ++j) {
            int col = col0 + j * 16;
            float bv = bias[col];
            #pragma unroll
            for (int r = 0; r < 8; ++r) {
                int row = mrow0 + i * 16 + r;
                float v = acc[i][j][r] + bv;
                if (DO_TANH) v = tanhf(v);
                out[(size_t)row * ldout + col] = v;
            }
        }
}

extern "C" void kernel_launch(void* const* d_in, const int* in_sizes, int n_in,
                              void* d_out, int out_size, void* d_ws, size_t ws_size,
                              hipStream_t stream) {
    (void)in_sizes; (void)n_in; (void)out_size; (void)ws_size;
    const int Brows = 8192, OBS = 256, NH = 2048, LAT2 = 256, KC = OBS + NH; // 2304

    const float* x     = (const float*)d_in[0];
    const float* h     = (const float*)d_in[1];
    const float* W_i2h = (const float*)d_in[2];   // [2048, 2304]
    const float* b_i2h = (const float*)d_in[3];
    const float* W_h2o = (const float*)d_in[4];   // [256, 2048]
    const float* b_h2o = (const float*)d_in[5];

    float* out   = (float*)d_out;                  // [8192, 256]
    float* h_new = out + (size_t)Brows * LAT2;     // [8192, 2048]

    // workspace layout (bf16 hi/lo planes)
    char* w = (char*)d_ws;
    unsigned short* cmb_hi = (unsigned short*)(w);                           // [8192,2304]
    unsigned short* cmb_lo = cmb_hi + (size_t)Brows * KC;
    unsigned short* wi_hi  = cmb_lo + (size_t)Brows * KC;                    // [2048,2304]
    unsigned short* wi_lo  = wi_hi  + (size_t)NH * KC;
    unsigned short* wo_hi  = wi_lo  + (size_t)NH * KC;                       // [256,2048]
    unsigned short* wo_lo  = wo_hi  + (size_t)LAT2 * NH;
    unsigned short* hn_hi  = wo_lo  + (size_t)LAT2 * NH;                     // [8192,2048]
    unsigned short* hn_lo  = hn_hi  + (size_t)Brows * NH;

    dim3 blk(256);
    dim3 cvt(2048);

    // pre-split inputs into bf16 hi/lo planes
    split_convert<<<cvt, blk, 0, stream>>>(x,     OBS, cmb_hi, cmb_lo, KC, 0,   Brows, OBS);
    split_convert<<<cvt, blk, 0, stream>>>(h,     NH,  cmb_hi, cmb_lo, KC, OBS, Brows, NH);
    split_convert<<<cvt, blk, 0, stream>>>(W_i2h, KC,  wi_hi,  wi_lo,  KC, 0,   NH,    KC);
    split_convert<<<cvt, blk, 0, stream>>>(W_h2o, NH,  wo_hi,  wo_lo,  NH, 0,   LAT2,  NH);

    // GEMM1: h_new = tanh([x|h] @ W_i2h^T + b_i2h)
    dim3 g1(Brows / BM, NH / BN);
    wmma_gemm<true><<<g1, blk, 0, stream>>>(cmb_hi, cmb_lo, KC,
                                            wi_hi,  wi_lo,  KC,
                                            b_i2h, h_new, NH);

    // split h_new for GEMM2
    split_convert<<<cvt, blk, 0, stream>>>(h_new, NH, hn_hi, hn_lo, NH, 0, Brows, NH);

    // GEMM2: out = h_new @ W_h2o^T + b_h2o
    dim3 g2(Brows / BM, LAT2 / BN);
    wmma_gemm<false><<<g2, blk, 0, stream>>>(hn_hi, hn_lo, NH,
                                             wo_hi, wo_lo, NH,
                                             b_h2o, out, LAT2);
}
